// ChebnetClassifier_78795470012462
// MI455X (gfx1250) — matrix operand order repaired and checked by reading, not compiled, not run on hardware
//
#include <hip/hip_runtime.h>

typedef __attribute__((ext_vector_type(16))) _Float16 v16h;
typedef __attribute__((ext_vector_type(8)))  float    v8f;

// ---------------- problem constants (from reference) ----------------
#define N0 200000
#define N1 50000
#define N2 12500
#define N2P 12512          // N2 padded to multiple of 16 for WMMA tiles
#define E0 (N0 * 32)
#define E1 (N1 * 32)
#define E2 (N2 * 32)
#define C_IN0 3
#define C_OUT0 64
#define C_OUT1 128
#define C_OUT2 256
#define KCHEB 6
#define FANIN 4
#define NCLS 10

// ---------------- elementwise helpers ----------------
__global__ void fill_f32(float* __restrict__ p, float v, int n) {
  int i = blockIdx.x * blockDim.x + threadIdx.x;
  if (i < n) p[i] = v;
}

__global__ void neg_copy_f32(float* __restrict__ dst, const float* __restrict__ src, int n) {
  int i = blockIdx.x * blockDim.x + threadIdx.x;
  if (i < n) dst[i] = -src[i];
}

// ---------------- graph normalization ----------------
__global__ void deg_kernel(const int* __restrict__ row, const int* __restrict__ col,
                           float* __restrict__ deg, int E) {
  int e = blockIdx.x * blockDim.x + threadIdx.x;
  if (e < E) {
    int r = row[e];
    if (r != col[e]) atomicAdd(&deg[r], 1.0f);
  }
}

__global__ void norm_kernel(const int* __restrict__ row, const int* __restrict__ col,
                            const float* __restrict__ deg, float* __restrict__ norm, int E) {
  int e = blockIdx.x * blockDim.x + threadIdx.x;
  if (e < E) {
    int r = row[e], c = col[e];
    float v = 0.0f;
    if (r != c) {
      float dr = deg[r], dc = deg[c];
      float ir = (dr > 0.0f) ? rsqrtf(dr) : 0.0f;
      float ic = (dc > 0.0f) ? rsqrtf(dc) : 0.0f;
      v = -ir * ic;
    }
    norm[e] = v;
  }
}

// out[row[e], :] += scale * norm[e] * h[col[e], :]
// grid.y = channel-group (4 channels each), grid.x covers edges
__global__ void spmm_scatter(float* __restrict__ out, const float* __restrict__ h,
                             const float* __restrict__ norm,
                             const int* __restrict__ row, const int* __restrict__ col,
                             int E, int C, float scale) {
  int e = blockIdx.x * blockDim.x + threadIdx.x;
  if (e >= E) return;
  float w = scale * norm[e];
  if (w == 0.0f) return;
  int r = row[e], c = col[e];
  int c0 = blockIdx.y << 2;
  const float* src = h + (long)c * C;
  float* dst = out + (long)r * C;
#pragma unroll
  for (int j = 0; j < 4; ++j) {
    int ch = c0 + j;
    if (ch < C) atomicAdd(&dst[ch], w * src[ch]);
  }
}

// ---------------- weight pre-pack into WMMA B-fragment order (f16) ----------------
// dst layout: [k_cheb][kb = K/32][nt = N/16][lane 0..31][j 0..15] contiguous halves.
// Lane (lg,lr): element j maps to B[kb*32 + (j<8 ? 8*lg + j : 16 + 8*lg + j - 8)][nt*16 + lr]; zero-pad kloc >= cin.
__global__ void pack_w_f16(_Float16* __restrict__ dst, const float* __restrict__ w,
                           int cin, int Kpad, int N) {
  int t = blockIdx.x * blockDim.x + threadIdx.x;
  int NT = N >> 4, KK = Kpad >> 5;
  int total = KCHEB * Kpad * N;
  if (t >= total) return;
  int j = t & 15;
  int rest = t >> 4;
  int lane = rest & 31;
  rest >>= 5;
  int nt = rest % NT; rest /= NT;
  int kb = rest % KK; int k = rest / KK;
  int lg = lane >> 4, lr = lane & 15;
  int kloc = kb * 32 + ((j < 8) ? (8 * lg + j) : (16 + 8 * lg + (j - 8)));
  int n = nt * 16 + lr;
  float v = (kloc < cin) ? w[((long)k * cin + kloc) * N + n] : 0.0f;
  dst[t] = (_Float16)v;
}

// ---------------- WMMA GEMM: out[Mp,N] += A[Mp,K] @ B[K,N] ----------------
// fp32 in/out, f16 MACs. One wave computes a 16x64 output strip (4 n-tiles).
// Mp multiple of 16, K multiple of 32, N multiple of 64. No bounds checks:
// everything padded in workspace, so all loads are unconditional vector loads.
__global__ void __launch_bounds__(32) wmma_gemm_acc(
    float* __restrict__ out, const float* __restrict__ A,
    const _Float16* __restrict__ BP, int N, int K) {
  const int m0 = blockIdx.x << 4;
  const int nt0 = blockIdx.y << 2;        // first of 4 n-tiles
  const int n0 = nt0 << 4;
  const int lane = threadIdx.x & 31;
  const int lg = lane >> 4;
  const int lr = lane & 15;
  const int NT = N >> 4;

  // C/D f32 16x16 layout: VGPR r holds M = m0 + r + 8*lg, N-col = n0 + lr (+16*t)
  v8f c0, c1, c2, c3;
#pragma unroll
  for (int r = 0; r < 8; ++r) {
    const float* cp = out + (long)(m0 + r + 8 * lg) * N + n0 + lr;
    c0[r] = cp[0];
    c1[r] = cp[16];
    c2[r] = cp[32];
    c3[r] = cp[48];
  }

  const float* arow = A + (long)(m0 + lr) * K;
  const int KK = K >> 5;
  for (int kb = 0; kb < KK; ++kb) {
    // A 16x32 f16 fragment: lane (lg,lr) holds k = kb*32 + {8lg..8lg+7, 16+8lg..16+8lg+7}
    const float4* p0 = (const float4*)(arow + (kb << 5) + 8 * lg);
    const float4* p1 = (const float4*)(arow + (kb << 5) + 16 + 8 * lg);
    float4 x0 = p0[0], x1 = p0[1];
    float4 y0 = p1[0], y1 = p1[1];
    v16h a;
    a[0] = (_Float16)x0.x; a[1] = (_Float16)x0.y; a[2]  = (_Float16)x0.z; a[3]  = (_Float16)x0.w;
    a[4] = (_Float16)x1.x; a[5] = (_Float16)x1.y; a[6]  = (_Float16)x1.z; a[7]  = (_Float16)x1.w;
    a[8] = (_Float16)y0.x; a[9] = (_Float16)y0.y; a[10] = (_Float16)y0.z; a[11] = (_Float16)y0.w;
    a[12] = (_Float16)y1.x; a[13] = (_Float16)y1.y; a[14] = (_Float16)y1.z; a[15] = (_Float16)y1.w;

    // B fragments: pre-packed, each lane reads 16 contiguous halves (32B)
    const _Float16* bbase = BP + (((long)kb * NT + nt0) << 9) + (lane << 4);
    v16h b0 = *(const v16h*)(bbase);
    v16h b1 = *(const v16h*)(bbase + 512);
    v16h b2 = *(const v16h*)(bbase + 1024);
    v16h b3 = *(const v16h*)(bbase + 1536);

    c0 = __builtin_amdgcn_wmma_f32_16x16x32_f16(false, a, false, b0, (short)0, c0, false, false);
    c1 = __builtin_amdgcn_wmma_f32_16x16x32_f16(false, a, false, b1, (short)0, c1, false, false);
    c2 = __builtin_amdgcn_wmma_f32_16x16x32_f16(false, a, false, b2, (short)0, c2, false, false);
    c3 = __builtin_amdgcn_wmma_f32_16x16x32_f16(false, a, false, b3, (short)0, c3, false, false);
  }

#pragma unroll
  for (int r = 0; r < 8; ++r) {
    float* cp = out + (long)(m0 + r + 8 * lg) * N + n0 + lr;
    cp[0]  = c0[r];
    cp[16] = c1[r];
    cp[32] = c2[r];
    cp[48] = c3[r];
  }
}

// ---------------- level-0 dense part: out[M,64] += A[M,3] @ B[3,64] ----------------
// K=3 makes WMMA wasteful; this is memory-bound, 3 FMAs per output element.
__global__ void gemm3_acc(float* __restrict__ out, const float* __restrict__ A,
                          const float* __restrict__ Bw, int M) {
  int t = blockIdx.x * blockDim.x + threadIdx.x;
  if (t >= M * 64) return;
  int m = t >> 6;
  int n = t & 63;
  const float* ar = A + m * 3;
  out[t] += ar[0] * Bw[n] + ar[1] * Bw[64 + n] + ar[2] * Bw[128 + n];
}

// ---------------- bias + optional relu (N is a power of two here) ----------------
__global__ void bias_act(float* __restrict__ out, const float* __restrict__ bias,
                         int total, int nmask, int relu) {
  int t = blockIdx.x * blockDim.x + threadIdx.x;
  if (t >= total) return;
  float v = out[t] + bias[t & nmask];
  if (relu) v = fmaxf(v, 0.0f);
  out[t] = v;
}

// ---------------- pooling: out[r,c] = mean_{f<4} in[cols[4r+f], c] ----------------
__global__ void pool_kernel(float* __restrict__ out, const float* __restrict__ in,
                            const int* __restrict__ cols, int n_out, int C) {
  int t = blockIdx.x * blockDim.x + threadIdx.x;
  if (t >= n_out * C) return;
  int r = t / C;
  int ch = t - r * C;
  const int* cc = cols + r * FANIN;
  float s = 0.0f;
#pragma unroll
  for (int f = 0; f < FANIN; ++f) s += in[(long)cc[f] * C + ch];
  out[t] = 0.25f * s;
}

// ---------------- final linear ----------------
__global__ void linear_init(float* __restrict__ out, const float* __restrict__ lb, int n) {
  int i = blockIdx.x * blockDim.x + threadIdx.x;
  if (i < n) out[i] = lb[i];
}

__global__ void linear_dot(float* __restrict__ out, const float* __restrict__ lw,
                           const float* __restrict__ h, long L) {
  int cls = blockIdx.y;
  long base = (long)blockIdx.x * 4096;
  const float* w = lw + (long)cls * L;
  float s = 0.0f;
  for (int i = threadIdx.x; i < 4096; i += 256) {
    long idx = base + i;
    if (idx < L) s += w[idx] * h[idx];
  }
  __shared__ float red[256];
  red[threadIdx.x] = s;
  __syncthreads();
  for (int o = 128; o > 0; o >>= 1) {
    if (threadIdx.x < o) red[threadIdx.x] += red[threadIdx.x + o];
    __syncthreads();
  }
  if (threadIdx.x == 0) atomicAdd(&out[cls], red[0]);
}

// ---------------- host-side enqueue helpers ----------------
static inline int cdiv(int a, int b) { return (a + b - 1) / b; }

static void launch_fill(float* p, float v, int n, hipStream_t s) {
  fill_f32<<<cdiv(n, 256), 256, 0, s>>>(p, v, n);
}

static void launch_scatter(float* out, const float* h, const float* norm,
                           const int* row, const int* col, int E, int C, float scale,
                           hipStream_t s) {
  dim3 grid(cdiv(E, 256), (C + 3) / 4);
  spmm_scatter<<<grid, 256, 0, s>>>(out, h, norm, row, col, E, C, scale);
}

// Full ChebConv (K=6) + bias (+relu) at one level.
// wpacked != nullptr -> WMMA path (cin in {64,128}); else gemm3 path (cin==3).
// Mp: row-padded node count (pad rows of all Tx buffers stay zero).
static void cheb_conv(const float* x, const int* eidx, int E, int Nn, int Mp,
                      int cin, int cout, const float* w, const _Float16* wpacked,
                      const float* b, float* out,
                      float* deg, float* norm, float* txA, float* txB, float* txC,
                      int relu, hipStream_t s) {
  const int* row = eidx;
  const int* col = eidx + E;

  launch_fill(deg, 0.0f, Nn, s);
  deg_kernel<<<cdiv(E, 256), 256, 0, s>>>(row, col, deg, E);
  norm_kernel<<<cdiv(E, 256), 256, 0, s>>>(row, col, deg, norm, E);

  dim3 ggrid(Mp / 16, cout / 64);
  auto gemm = [&](const float* Amat, int k) {
    if (wpacked) {
      wmma_gemm_acc<<<ggrid, 32, 0, s>>>(out, Amat, wpacked + (long)k * cin * cout,
                                         cout, cin);
    } else {
      gemm3_acc<<<cdiv(Mp * 64, 256), 256, 0, s>>>(out, Amat, w + k * cin * cout, Mp);
    }
  };

  launch_fill(out, 0.0f, Mp * cout, s);
  gemm(x, 0);

  // Tx1 = prop(x)
  launch_fill(txB, 0.0f, Mp * cin, s);
  launch_scatter(txB, x, norm, row, col, E, cin, 1.0f, s);
  gemm(txB, 1);

  // Tx2 = 2*prop(Tx1) - Tx0, rotating through 3 scratch buffers
  const float* Tx0 = x;
  float* Tx1 = txB;
  float* Tx2 = txA;
  for (int k = 2; k < KCHEB; ++k) {
    neg_copy_f32<<<cdiv(Mp * cin, 256), 256, 0, s>>>(Tx2, Tx0, Mp * cin);
    launch_scatter(Tx2, Tx1, norm, row, col, E, cin, 2.0f, s);
    gemm(Tx2, k);
    const float* oldTx0 = Tx0;
    Tx0 = Tx1;
    Tx1 = Tx2;
    Tx2 = (k == 2) ? txC : (float*)oldTx0;
  }

  bias_act<<<cdiv(Nn * cout, 256), 256, 0, s>>>(out, b, Nn * cout, cout - 1, relu);
}

extern "C" void kernel_launch(void* const* d_in, const int* in_sizes, int n_in,
                              void* d_out, int out_size, void* d_ws, size_t ws_size,
                              hipStream_t stream) {
  const float* x   = (const float*)d_in[0];
  const int*   ei0 = (const int*)d_in[1];
  const int*   ei1 = (const int*)d_in[2];
  const int*   ei2 = (const int*)d_in[3];
  const int*   pc0 = (const int*)d_in[4];
  const int*   pc1 = (const int*)d_in[5];
  const float* w0  = (const float*)d_in[6];
  const float* b0  = (const float*)d_in[7];
  const float* w1  = (const float*)d_in[8];
  const float* b1  = (const float*)d_in[9];
  const float* w2  = (const float*)d_in[10];
  const float* b2  = (const float*)d_in[11];
  const float* lw  = (const float*)d_in[12];
  const float* lb  = (const float*)d_in[13];
  float* out = (float*)d_out;

  // workspace carve-up (float offsets; all 16B aligned)
  float* ws   = (float*)d_ws;
  float* DEG  = ws;                         // 200,000
  float* NORM = DEG + N0;                   // 6,400,000
  float* TXA  = NORM + E0;                  // 3,200,000 each
  float* TXB  = TXA + 3200000;
  float* TXC  = TXB + 3200000;
  float* OUT  = TXC + 3200000;              // 12,800,000 (reused across levels)
  float* H1   = OUT + (long)N0 * C_OUT0;    // 3,200,000
  float* H2   = H1 + (long)N1 * C_OUT0;     // 12512*128 = 1,601,536
  _Float16* WP1 = (_Float16*)(H2 + (long)N2P * C_OUT1);  // 49,152 halves
  _Float16* WP2 = WP1 + KCHEB * C_OUT0 * C_OUT1;         // 196,608 halves

  // pre-pack level-1/2 weights into WMMA B-fragment f16 layout
  {
    int t1 = KCHEB * C_OUT0 * C_OUT1;
    pack_w_f16<<<cdiv(t1, 256), 256, 0, stream>>>(WP1, w1, C_OUT0, C_OUT0, C_OUT1);
    int t2 = KCHEB * C_OUT1 * C_OUT2;
    pack_w_f16<<<cdiv(t2, 256), 256, 0, stream>>>(WP2, w2, C_OUT1, C_OUT1, C_OUT2);
  }

  // level 0: ChebConv(3 -> 64) + relu on 200k nodes (non-WMMA dense part)
  cheb_conv(x, ei0, E0, N0, N0, C_IN0, C_OUT0, w0, nullptr, b0, OUT,
            DEG, NORM, TXA, TXB, TXC, /*relu=*/1, stream);
  // pool0 -> H1 [50000 x 64]
  pool_kernel<<<cdiv(N1 * C_OUT0, 256), 256, 0, stream>>>(H1, OUT, pc0, N1, C_OUT0);

  // level 1: ChebConv(64 -> 128) + relu on 50k nodes (WMMA)
  cheb_conv(H1, ei1, E1, N1, N1, C_OUT0, C_OUT1, w1, WP1, b1, OUT,
            DEG, NORM, TXA, TXB, TXC, /*relu=*/1, stream);
  // pool1 -> H2 [12512(pad) x 128], pad rows zeroed
  launch_fill(H2, 0.0f, N2P * C_OUT1, stream);
  pool_kernel<<<cdiv(N2 * C_OUT1, 256), 256, 0, stream>>>(H2, OUT, pc1, N2, C_OUT1);

  // level 2: ChebConv(128 -> 256), no relu, rows padded to 12512 (WMMA)
  cheb_conv(H2, ei2, E2, N2, N2P, C_OUT1, C_OUT2, w2, WP2, b2, OUT,
            DEG, NORM, TXA, TXB, TXC, /*relu=*/0, stream);

  // classifier: out = lb + lw @ flatten(OUT[0:12500, :])
  const long L = (long)N2 * C_OUT2;  // 3,200,000 (pad rows are past this range)
  linear_init<<<1, 32, 0, stream>>>(out, lb, NCLS);
  dim3 lgrid((unsigned)((L + 4095) / 4096), NCLS);
  linear_dot<<<lgrid, 256, 0, stream>>>(out, lw, OUT, L);
}
